// CharRep_7189775254032
// MI455X (gfx1250) — matrix operand order repaired and checked by reading, not compiled, not run on hardware
//
#include <hip/hip_runtime.h>

typedef __attribute__((ext_vector_type(16))) _Float16 v16h;
typedef __attribute__((ext_vector_type(8)))  _Float16 v8h;
typedef __attribute__((ext_vector_type(8)))  float    v8f;

#define NSEQ_TOT 16384   // B*S = 64*256
#define TT       16      // chars per token
#define E        30      // char emb
#define EP       32      // padded K for input proj
#define H        50      // hidden per direction
#define HP       64      // padded hidden (K for recurrence, per-gate col pad)
#define GCOLS    256     // 4 gates * 64 padded columns
#define SPW      16      // sequences per workgroup (WMMA M)
#define THREADS  128     // 4 waves of 32

__device__ __forceinline__ float fast_tanh(float x) {
#if __has_builtin(__builtin_amdgcn_tanhf)
    return __builtin_amdgcn_tanhf(x);          // gfx1250 v_tanh_f32 (TRANS unit)
#elif __has_builtin(__builtin_amdgcn_tanh_f32)
    return __builtin_amdgcn_tanh_f32(x);
#else
    float ax = fabsf(x);
    float e  = __expf(-2.0f * ax);
    return copysignf((1.0f - e) / (1.0f + e), x);
#endif
}
__device__ __forceinline__ float fast_sig(float x) {
    return 0.5f + 0.5f * fast_tanh(0.5f * x);  // sigmoid via single TRANS op
}

__global__ __launch_bounds__(THREADS) void char_bilstm_kernel(
    const int*   __restrict__ ids,       // [NSEQ_TOT, TT]
    const float* __restrict__ emb,       // [100, E]
    const float* __restrict__ wih_f, const float* __restrict__ whh_f,
    const float* __restrict__ bih_f, const float* __restrict__ bhh_f,
    const float* __restrict__ wih_r, const float* __restrict__ whh_r,
    const float* __restrict__ bih_r, const float* __restrict__ bhh_r,
    float*       __restrict__ out)       // [NSEQ_TOT, 100]
{
    __shared__ _Float16 x_s[TT][SPW][EP];      // 16 KB  (f16 embeddings, K-padded)
    __shared__ float    gates_s[SPW][GCOLS];   // 16 KB  (pre-activation gates)
    __shared__ _Float16 h_s[SPW][HP];          // 2 KB   (hidden state, f16, padded)
    __shared__ int      ids_s[SPW][TT];
    __shared__ int      tsel_s[SPW];           // max(len-1, 0)

    const int tid  = threadIdx.x;
    const int lane = tid & 31;
    const int wave = tid >> 5;                 // wave == gate index (i,f,g,o)
    const int seq0 = blockIdx.x * SPW;

    // ---------- stage char ids ----------
    for (int k = tid; k < SPW * TT; k += THREADS) {
        int s = k / TT, t = k % TT;
        ids_s[s][t] = ids[(seq0 + s) * TT + t];
    }
    __syncthreads();
    if (tid < SPW) {
        int len = 0;
        for (int t = 0; t < TT; ++t) len += (ids_s[tid][t] != 0) ? 1 : 0;
        tsel_s[tid] = (len - 1 > 0) ? (len - 1) : 0;
    }
    // ---------- stage embeddings f32->f16, zero-pad K 30..31 ----------
    for (int idx = tid; idx < TT * SPW * EP; idx += THREADS) {
        int k = idx & (EP - 1);
        int s = (idx >> 5) & (SPW - 1);
        int t = idx >> 9;
        int id = ids_s[s][t];
        float v = (k < E) ? emb[id * E + k] : 0.0f;
        x_s[t][s][k] = (_Float16)v;
    }
    __syncthreads();

    const int eseq  = tid >> 3;                // elementwise: sequence 0..15
    const int ehid0 = (tid & 7) * 8;           // elementwise: 8 hid slots

    for (int dir = 0; dir < 2; ++dir) {
        const float* wih = dir ? wih_r : wih_f;
        const float* whh = dir ? whh_r : whh_f;
        const float* bih = dir ? bih_r : bih_f;
        const float* bhh = dir ? bhh_r : bhh_f;

        // ---- gather weight B-operands into VGPRs (ISA B layout:
        //      N = lane%16, K = (lane/16)*8 + e (+8 for e>=8)), zero-padded ----
        const int ncol = lane & 15;
        const int khi  = (lane >> 4) * 8;
        v16h bih_op[4];
        v16h bhh_op[4][2];
        float bias_v[4];
        for (int j = 0; j < 4; ++j) {
            int c   = j * 16 + ncol;           // column within this gate, 0..63
            int row = wave * H + c;            // row in [4H, *] weight matrices
            bool cv = (c < H);
            v16h bi, bh0, bh1;
            for (int e = 0; e < 16; ++e) {
                int k  = khi + e + ((e >= 8) ? 8 : 0);
                int k2 = 32 + k;
                bi[e]  = (cv && k  < E) ? (_Float16)wih[row * E + k]  : (_Float16)0.0f;
                bh0[e] = (cv && k  < H) ? (_Float16)whh[row * H + k]  : (_Float16)0.0f;
                bh1[e] = (cv && k2 < H) ? (_Float16)whh[row * H + k2] : (_Float16)0.0f;
            }
            bih_op[j]    = bi;
            bhh_op[j][0] = bh0;
            bhh_op[j][1] = bh1;
            bias_v[j]    = cv ? (bih[row] + bhh[row]) : 0.0f;
        }

        // ---- reset state ----
        float c_reg[8];
        #pragma unroll
        for (int j = 0; j < 8; ++j) c_reg[j] = 0.0f;
        {
            v8h z;
            #pragma unroll
            for (int j = 0; j < 8; ++j) z[j] = (_Float16)0.0f;
            *(v8h*)&h_s[eseq][ehid0] = z;
        }
        __syncthreads();

        const int arow = lane & 15;            // A layout: M = lane%16
        const int k0a  = (lane < 16) ? 0 : 8;  // A layout: K group per lane half

        for (int t = 0; t < TT; ++t) {
            int xt = dir ? (TT - 1 - t) : t;   // reversed scan for backward dir

            // A operand: x_t  (16x32 f16)
            v16h ax;
            {
                v8h lo = *(const v8h*)&x_s[xt][arow][k0a];
                v8h hi = *(const v8h*)&x_s[xt][arow][16 + k0a];
                #pragma unroll
                for (int e = 0; e < 8; ++e) { ax[e] = lo[e]; ax[8 + e] = hi[e]; }
            }
            // A operands: h (16x64 f16 -> two 16x32 K-chunks)
            v16h ah0, ah1;
            {
                v8h lo0 = *(const v8h*)&h_s[arow][k0a];
                v8h hi0 = *(const v8h*)&h_s[arow][16 + k0a];
                v8h lo1 = *(const v8h*)&h_s[arow][32 + k0a];
                v8h hi1 = *(const v8h*)&h_s[arow][48 + k0a];
                #pragma unroll
                for (int e = 0; e < 8; ++e) {
                    ah0[e] = lo0[e]; ah0[8 + e] = hi0[e];
                    ah1[e] = lo1[e]; ah1[8 + e] = hi1[e];
                }
            }

            // gates[16, 64] for this wave's gate: bias + x@Wih^T + h@Whh^T
            const int rbase = (lane >> 4) * 8; // C/D layout: M = r + 8*(lane/16)
            #pragma unroll
            for (int j = 0; j < 4; ++j) {
                v8f acc;
                #pragma unroll
                for (int r = 0; r < 8; ++r) acc[r] = bias_v[j];
                acc = __builtin_amdgcn_wmma_f32_16x16x32_f16(
                          false, ax,  false, bih_op[j],    (short)0, acc, false, false);
                acc = __builtin_amdgcn_wmma_f32_16x16x32_f16(
                          false, ah0, false, bhh_op[j][0], (short)0, acc, false, false);
                acc = __builtin_amdgcn_wmma_f32_16x16x32_f16(
                          false, ah1, false, bhh_op[j][1], (short)0, acc, false, false);
                int col = wave * HP + j * 16 + ncol;
                #pragma unroll
                for (int r = 0; r < 8; ++r)
                    gates_s[rbase + r][col] = acc[r];
            }
            __syncthreads();

            // ---- LSTM cell update: fully unrolled, vector LDS access ----
            {
                v8f gi = *(const v8f*)&gates_s[eseq][ehid0];
                v8f gf = *(const v8f*)&gates_s[eseq][HP + ehid0];
                v8f gg = *(const v8f*)&gates_s[eseq][2 * HP + ehid0];
                v8f go = *(const v8f*)&gates_s[eseq][3 * HP + ehid0];
                float hval[8];
                v8h  hv;
                #pragma unroll
                for (int j = 0; j < 8; ++j) {
                    float cc = fast_sig(gf[j]) * c_reg[j]
                             + fast_sig(gi[j]) * fast_tanh(gg[j]);
                    float hh = fast_sig(go[j]) * fast_tanh(cc);
                    c_reg[j] = cc;
                    hval[j]  = hh;
                    hv[j]    = ((ehid0 + j) < H) ? (_Float16)hh : (_Float16)0.0f;
                }
                *(v8h*)&h_s[eseq][ehid0] = hv;

                const bool do_fwd = (dir == 0) && (t == tsel_s[eseq]);
                const bool do_bwd = (dir == 1) && (t == TT - 1);
                if (do_fwd) {
                    #pragma unroll
                    for (int j = 0; j < 8; ++j) {
                        int hid = ehid0 + j;
                        if (hid < H) out[(seq0 + eseq) * 100 + hid] = hval[j];
                    }
                } else if (do_bwd) {
                    #pragma unroll
                    for (int j = 0; j < 8; ++j) {
                        int hid = ehid0 + j;
                        if (hid < H) out[(seq0 + eseq) * 100 + H + hid] = hval[j];
                    }
                }
            }
            __syncthreads();
        }
    }
}

extern "C" void kernel_launch(void* const* d_in, const int* in_sizes, int n_in,
                              void* d_out, int out_size, void* d_ws, size_t ws_size,
                              hipStream_t stream) {
    const int*   ids   = (const int*)  d_in[0];
    const float* emb   = (const float*)d_in[1];
    const float* wih_f = (const float*)d_in[2];
    const float* whh_f = (const float*)d_in[3];
    const float* bih_f = (const float*)d_in[4];
    const float* bhh_f = (const float*)d_in[5];
    const float* wih_r = (const float*)d_in[6];
    const float* whh_r = (const float*)d_in[7];
    const float* bih_r = (const float*)d_in[8];
    const float* bhh_r = (const float*)d_in[9];
    float* out = (float*)d_out;

    dim3 grid(NSEQ_TOT / SPW);   // 1024 workgroups, 16 sequences each
    char_bilstm_kernel<<<grid, THREADS, 0, stream>>>(
        ids, emb, wih_f, whh_f, bih_f, bhh_f,
        wih_r, whh_r, bih_r, bhh_r, out);
}
